// SpectralGate_91018946936966
// MI455X (gfx1250) — compile-verified, matrix operand order checked by reading, not run on hardware
//
#include <hip/hip_runtime.h>
#include <cstdint>
#include <cstddef>

// ---------------------------------------------------------------------------
// SpectralGate for MI455X (gfx1250, wave32, WMMA).
// Pipeline: rfft2 -> LN -> Linear -> GELU -> Linear -> residual -> irfft2.
// HBM-bound (~2.2 GB moved); GEMMs done with v_wmma_f32_16x16x32_bf16.
// ---------------------------------------------------------------------------

typedef __attribute__((ext_vector_type(16))) __bf16 v16bf;
typedef __attribute__((ext_vector_type(8)))  float  v8f;

#define Bn 8
#define Cc 128
#define Hh 256
#define Wfull 256
#define Wf 129
#define Dd 256          // 2*C feature dim
#define MROWS 64        // rows per MLP block

static __device__ __forceinline__ unsigned short f2bf(float f) {
  unsigned u = __builtin_bit_cast(unsigned, f);
  unsigned r = 0x7FFFu + ((u >> 16) & 1u);       // round-to-nearest-even
  return (unsigned short)((u + r) >> 16);
}

// ---------------------------------------------------------------------------
// 256-point radix-2 Stockham FFT in LDS. 128 threads per transform, each does
// one butterfly per stage. re/im are [2][256] ping-pong buffers; result lands
// in buffer 0 (8 stages toggle back). sgn = -1 forward, +1 inverse.
// ---------------------------------------------------------------------------
static __device__ __forceinline__ void fft256_stages(float (*re)[256],
                                                     float (*im)[256],
                                                     int j, float sgn) {
  int src = 0;
  #pragma unroll
  for (int Ns = 1; Ns < 256; Ns <<= 1) {
    __syncthreads();
    int m    = j & (Ns - 1);
    int base = ((j - m) << 1) + m;               // (j/Ns)*2Ns + m
    float ang = sgn * 3.14159265358979323846f * (float)m / (float)Ns;
    float sn, cs;
    __sincosf(ang, &sn, &cs);
    float v0r = re[src][j],       v0i = im[src][j];
    float tr  = re[src][j + 128], ti  = im[src][j + 128];
    float v1r = tr * cs - ti * sn;
    float v1i = tr * sn + ti * cs;
    re[src ^ 1][base]      = v0r + v1r;
    im[src ^ 1][base]      = v0i + v1i;
    re[src ^ 1][base + Ns] = v0r - v1r;
    im[src ^ 1][base + Ns] = v0i - v1i;
    src ^= 1;
  }
  __syncthreads();
}

// ---- forward FFT along W for each (b,c,h) row; keep bins 0..128 ------------
__global__ void __launch_bounds__(256) fft_rows_fwd(const float* __restrict__ x,
                                                    float2* __restrict__ spec) {
  __shared__ float re[2][2][256], im[2][2][256];
  int rb = threadIdx.x >> 7;                     // row within block (2/block)
  int j  = threadIdx.x & 127;
  size_t row = (size_t)blockIdx.x * 2 + rb;      // (b*C + c)*H + h
  const float* xp = x + row * Wfull;
  float (*pre)[256] = re[rb];
  float (*pim)[256] = im[rb];
  pre[0][j]       = xp[j];
  pre[0][j + 128] = xp[j + 128];
  pim[0][j] = 0.f;  pim[0][j + 128] = 0.f;
  fft256_stages(pre, pim, j, -1.f);
  float2* op = spec + row * Wf;
  op[j] = make_float2(pre[0][j], pim[0][j]);
  if (j == 0) op[128] = make_float2(pre[0][128], pim[0][128]);
}

// ---- forward FFT along H + transpose-scatter into [B,H,Wf,2C], x 1/256 -----
__global__ void __launch_bounds__(256) fft_cols_fwd(const float2* __restrict__ spec,
                                                    float* __restrict__ flat) {
  __shared__ float re[2][2][256], im[2][2][256];
  int cb = threadIdx.x >> 7;
  int j  = threadIdx.x & 127;
  size_t col = (size_t)blockIdx.x * 2 + cb;      // (b*C + c)*Wf + wf
  int wf = (int)(col % Wf);
  int bc = (int)(col / Wf);
  int c  = bc & (Cc - 1);
  int b  = bc >> 7;
  const float2* ip = spec + (size_t)bc * (Hh * Wf) + wf;
  float (*pre)[256] = re[cb];
  float (*pim)[256] = im[cb];
  float2 v0 = ip[(size_t)j * Wf];
  float2 v1 = ip[(size_t)(j + 128) * Wf];
  pre[0][j] = v0.x;       pim[0][j] = v0.y;
  pre[0][j + 128] = v1.x; pim[0][j + 128] = v1.y;
  fft256_stages(pre, pim, j, -1.f);
  const float S = 1.f / 256.f;                   // ortho: 1/sqrt(H*W)
  #pragma unroll
  for (int t = 0; t < 2; ++t) {
    int h = j + t * 128;
    size_t fi = (((size_t)(b * Hh + h) * Wf + wf) * Dd) + 2 * c;
    *reinterpret_cast<float2*>(flat + fi) =
        make_float2(pre[0][h] * S, pim[0][h] * S);
  }
}

// ---------------------------------------------------------------------------
// Fused LayerNorm + MLP (WMMA bf16) + GELU + residual, in place on `flat`.
// 64 rows/block, 8 waves; wave w: M-subtile mt=w&3, N-tiles nt=(w>>2)+2i.
// Weights pre-swizzled: wsw[kt][nt][lane][16 bf16] = B-fragment per lane.
// ---------------------------------------------------------------------------
__global__ void __launch_bounds__(256) mlp_kernel(
    float* __restrict__ flat,
    const unsigned short* __restrict__ wsw1,
    const unsigned short* __restrict__ wsw2,
    const float* __restrict__ gamma,
    const float* __restrict__ beta,
    const float* __restrict__ b1,
    const float* __restrict__ b2,
    const float* __restrict__ scale_p) {
  __shared__ unsigned short Sbuf[MROWS][Dd + 8];   // +8 halves: de-bank-conflict

  const int tid = threadIdx.x;
  const size_t row0 = (size_t)blockIdx.x * MROWS;

  // -------- load + LayerNorm (4 threads / row, shfl reduction) --------
  {
    int r = tid >> 2, q = tid & 3;
    const float* rp = flat + (row0 + r) * Dd + q * 64;
    float v[64];
    float s = 0.f, s2 = 0.f;
    #pragma unroll
    for (int i = 0; i < 16; ++i) {
      float4 f4 = reinterpret_cast<const float4*>(rp)[i];
      v[4*i+0] = f4.x; v[4*i+1] = f4.y; v[4*i+2] = f4.z; v[4*i+3] = f4.w;
      s  += f4.x + f4.y + f4.z + f4.w;
      s2 += f4.x*f4.x + f4.y*f4.y + f4.z*f4.z + f4.w*f4.w;
    }
    s  += __shfl_xor(s, 1);  s  += __shfl_xor(s, 2);
    s2 += __shfl_xor(s2, 1); s2 += __shfl_xor(s2, 2);
    float mean = s * (1.f / 256.f);
    float var  = s2 * (1.f / 256.f) - mean * mean;
    float rstd = rsqrtf(var + 1e-5f);
    #pragma unroll
    for (int i = 0; i < 64; ++i) {
      int k = q * 64 + i;
      float nx = (v[i] - mean) * rstd * gamma[k] + beta[k];
      Sbuf[r][k] = f2bf(nx);
    }
  }
  __syncthreads();

  const int lane  = tid & 31;
  const int wave  = tid >> 5;
  const int mt    = wave & 3;       // M subtile (16 rows)
  const int nt0   = wave >> 2;      // N-tile stride-2 start
  const int mloc  = lane & 15;
  const int khalf = lane >> 4;      // K-half per 16-bit A/B layout

  union V16 { v16bf v; unsigned u[8]; };
  v8f acc[8];

  // ================= GEMM1: normed @ W1^T =================
  {
    __builtin_prefetch(wsw1 + (size_t)lane * 16, 0, 0);
    V16 a[8];
    #pragma unroll
    for (int kt = 0; kt < 8; ++kt) {
      const unsigned short* ar = &Sbuf[mt * 16 + mloc][kt * 32 + khalf * 8];
      #pragma unroll
      for (int g = 0; g < 8; ++g) {
        int off = ((g >> 2) << 4) + ((g & 3) << 1);   // K pattern within frag
        a[kt].u[g] = *reinterpret_cast<const unsigned*>(ar + off);
      }
    }
    #pragma unroll
    for (int i = 0; i < 8; ++i) {
      int nt = nt0 + 2 * i;
      v8f c = {0.f, 0.f, 0.f, 0.f, 0.f, 0.f, 0.f, 0.f};
      #pragma unroll
      for (int kt = 0; kt < 8; ++kt) {
        const uint4* bp = reinterpret_cast<const uint4*>(
            wsw1 + ((size_t)((kt * 16 + nt) * 32 + lane) << 4));
        V16 bb;
        uint4 lo = bp[0], hi = bp[1];
        bb.u[0]=lo.x; bb.u[1]=lo.y; bb.u[2]=lo.z; bb.u[3]=lo.w;
        bb.u[4]=hi.x; bb.u[5]=hi.y; bb.u[6]=hi.z; bb.u[7]=hi.w;
        c = __builtin_amdgcn_wmma_f32_16x16x32_bf16(false, a[kt].v, false, bb.v,
                                                    (short)0, c, false, false);
      }
      acc[i] = c;
    }
  }
  __syncthreads();   // all Sbuf(normed) reads complete before overwrite

  // -------- epilogue 1: bias + exact GELU -> bf16 hidden into Sbuf --------
  #pragma unroll
  for (int i = 0; i < 8; ++i) {
    int nt = nt0 + 2 * i;
    #pragma unroll
    for (int rr = 0; rr < 8; ++rr) {
      int m = mt * 16 + rr + 8 * khalf;   // C/D layout: VGPR rr, lane half
      int n = nt * 16 + mloc;
      float xx = acc[i][rr] + b1[n];
      xx = 0.5f * xx * (1.0f + erff(xx * 0.70710678118654752f));
      Sbuf[m][n] = f2bf(xx);
    }
  }
  __syncthreads();

  // ================= GEMM2: hidden @ W2^T =================
  {
    __builtin_prefetch(wsw2 + (size_t)lane * 16, 0, 0);
    V16 a[8];
    #pragma unroll
    for (int kt = 0; kt < 8; ++kt) {
      const unsigned short* ar = &Sbuf[mt * 16 + mloc][kt * 32 + khalf * 8];
      #pragma unroll
      for (int g = 0; g < 8; ++g) {
        int off = ((g >> 2) << 4) + ((g & 3) << 1);
        a[kt].u[g] = *reinterpret_cast<const unsigned*>(ar + off);
      }
    }
    #pragma unroll
    for (int i = 0; i < 8; ++i) {
      int nt = nt0 + 2 * i;
      v8f c = {0.f, 0.f, 0.f, 0.f, 0.f, 0.f, 0.f, 0.f};
      #pragma unroll
      for (int kt = 0; kt < 8; ++kt) {
        const uint4* bp = reinterpret_cast<const uint4*>(
            wsw2 + ((size_t)((kt * 16 + nt) * 32 + lane) << 4));
        V16 bb;
        uint4 lo = bp[0], hi = bp[1];
        bb.u[0]=lo.x; bb.u[1]=lo.y; bb.u[2]=lo.z; bb.u[3]=lo.w;
        bb.u[4]=hi.x; bb.u[5]=hi.y; bb.u[6]=hi.z; bb.u[7]=hi.w;
        c = __builtin_amdgcn_wmma_f32_16x16x32_bf16(false, a[kt].v, false, bb.v,
                                                    (short)0, c, false, false);
      }
      acc[i] = c;
    }
  }

  // -------- epilogue 2: bias + residual (in place) --------
  float sc = scale_p[0];
  #pragma unroll
  for (int i = 0; i < 8; ++i) {
    int nt = nt0 + 2 * i;
    #pragma unroll
    for (int rr = 0; rr < 8; ++rr) {
      int m = mt * 16 + rr + 8 * khalf;
      int n = nt * 16 + mloc;
      size_t gi = (row0 + m) * Dd + n;
      flat[gi] = flat[gi] + sc * (acc[i][rr] + b2[n]);
    }
  }
}

// ---- inverse FFT along H: gather from [B,H,Wf,2C] back to [B,C,H,Wf] -------
__global__ void __launch_bounds__(256) fft_cols_inv(const float* __restrict__ flat,
                                                    float2* __restrict__ spec) {
  __shared__ float re[2][2][256], im[2][2][256];
  int cb = threadIdx.x >> 7;
  int j  = threadIdx.x & 127;
  size_t col = (size_t)blockIdx.x * 2 + cb;
  int wf = (int)(col % Wf);
  int bc = (int)(col / Wf);
  int c  = bc & (Cc - 1);
  int b  = bc >> 7;
  float (*pre)[256] = re[cb];
  float (*pim)[256] = im[cb];
  #pragma unroll
  for (int t = 0; t < 2; ++t) {
    int h = j + t * 128;
    size_t fi = (((size_t)(b * Hh + h) * Wf + wf) * Dd) + 2 * c;
    float2 v = *reinterpret_cast<const float2*>(flat + fi);
    pre[0][h] = v.x;  pim[0][h] = v.y;
  }
  fft256_stages(pre, pim, j, +1.f);
  float2* op = spec + (size_t)bc * (Hh * Wf) + wf;
  op[(size_t)j * Wf]         = make_float2(pre[0][j], pim[0][j]);
  op[(size_t)(j + 128) * Wf] = make_float2(pre[0][j + 128], pim[0][j + 128]);
}

// ---- inverse FFT along W with Hermitian extension -> real out, x 1/256 -----
__global__ void __launch_bounds__(256) fft_rows_inv(const float2* __restrict__ spec,
                                                    float* __restrict__ out) {
  __shared__ float re[2][2][256], im[2][2][256];
  int rb = threadIdx.x >> 7;
  int j  = threadIdx.x & 127;
  size_t row = (size_t)blockIdx.x * 2 + rb;
  const float2* ip = spec + row * Wf;
  float (*pre)[256] = re[rb];
  float (*pim)[256] = im[rb];
  float2 v = ip[j];
  pre[0][j] = v.x;  pim[0][j] = v.y;
  float2 w = (j == 0) ? ip[128] : ip[128 - j];    // X[k]=conj(X[256-k])
  pre[0][j + 128] = w.x;
  pim[0][j + 128] = (j == 0) ? w.y : -w.y;
  fft256_stages(pre, pim, j, +1.f);
  const float S = 1.f / 256.f;                    // ortho inverse scale
  float* op = out + row * Wfull;
  op[j]       = pre[0][j] * S;
  op[j + 128] = pre[0][j + 128] * S;
}

// ---- f32 weights -> bf16, swizzled to per-lane WMMA B-fragment order -------
// Layout: uint t = kt*4096 + nt*256 + lane*8 + g ; pair (k, k+1) of W^T col n.
__global__ void __launch_bounds__(256) swizzle_w(const float* __restrict__ W,
                                                 unsigned* __restrict__ outw) {
  int t = blockIdx.x * blockDim.x + threadIdx.x;   // 0..32767
  int g    = t & 7;
  int lane = (t >> 3) & 31;
  int nt   = (t >> 8) & 15;
  int kt   = (t >> 12);
  int n = nt * 16 + (lane & 15);
  int k = kt * 32 + ((lane >> 4) << 3) + ((g >> 2) << 4) + ((g & 3) << 1);
  unsigned short h0 = f2bf(W[n * 256 + k]);        // B[k][n]   = W[n][k]
  unsigned short h1 = f2bf(W[n * 256 + k + 1]);    // B[k+1][n] = W[n][k+1]
  outw[t] = (unsigned)h0 | ((unsigned)h1 << 16);
}

// ---------------------------------------------------------------------------
extern "C" void kernel_launch(void* const* d_in, const int* in_sizes, int n_in,
                              void* d_out, int out_size, void* d_ws, size_t ws_size,
                              hipStream_t stream) {
  const float* x     = (const float*)d_in[0];
  const float* gamma = (const float*)d_in[1];
  const float* beta  = (const float*)d_in[2];
  const float* W1    = (const float*)d_in[3];
  const float* b1    = (const float*)d_in[4];
  const float* W2    = (const float*)d_in[5];
  const float* b2    = (const float*)d_in[6];
  const float* scale = (const float*)d_in[7];

  // workspace layout: [ flat f32 | spec float2 | wsw1 bf16 | wsw2 bf16 ]
  const size_t FLAT_ELEMS = (size_t)Bn * Hh * Wf * Dd;   // 67,633,152 f32
  const size_t SPEC_ELEMS = (size_t)Bn * Cc * Hh * Wf;   // 33,816,576 float2
  float*    flat = (float*)d_ws;
  float2*   spec = (float2*)((char*)d_ws + FLAT_ELEMS * sizeof(float));
  unsigned* wsw1 = (unsigned*)((char*)spec + SPEC_ELEMS * sizeof(float2));
  unsigned* wsw2 = wsw1 + 32768;

  swizzle_w<<<128, 256, 0, stream>>>(W1, wsw1);
  swizzle_w<<<128, 256, 0, stream>>>(W2, wsw2);

  fft_rows_fwd<<<(Bn * Cc * Hh) / 2, 256, 0, stream>>>(x, spec);
  fft_cols_fwd<<<(Bn * Cc * Wf) / 2, 256, 0, stream>>>(spec, flat);

  mlp_kernel<<<(Bn * Hh * Wf) / MROWS, 256, 0, stream>>>(
      flat, (const unsigned short*)wsw1, (const unsigned short*)wsw2,
      gamma, beta, b1, b2, scale);

  fft_cols_inv<<<(Bn * Cc * Wf) / 2, 256, 0, stream>>>(flat, spec);
  fft_rows_inv<<<(Bn * Cc * Hh) / 2, 256, 0, stream>>>(spec, (float*)d_out);
}
